// Seq2Seq_30597347017226
// MI455X (gfx1250) — matrix operand order repaired and checked by reading, not compiled, not run on hardware
//
#include <hip/hip_runtime.h>
#include <hip/hip_bf16.h>
#include <stdint.h>

// ---------------------------------------------------------------------------
// Types for CDNA5 (gfx1250) WMMA: bf16 16x16x32, f32 accumulate.
// ---------------------------------------------------------------------------
typedef __bf16 bf16_t;
typedef __attribute__((ext_vector_type(16))) __bf16 v16bf;
typedef __attribute__((ext_vector_type(8)))  __bf16 v8bf;
typedef __attribute__((ext_vector_type(8)))  float  v8f;
typedef __attribute__((ext_vector_type(4)))  int    v4i;

#define WMMA_BF16(a, b, c) \
  __builtin_amdgcn_wmma_f32_16x16x32_bf16(false, (a), false, (b), (short)0, (c), false, false)

// gfx1250 async global->LDS path (ASYNCcnt-tracked). Probe-confirmed present
// in this toolchain; parameter types are v4i* in AS1 (global) / AS3 (LDS).
#if defined(__HIP_DEVICE_COMPILE__) && \
    __has_builtin(__builtin_amdgcn_global_load_async_to_lds_b128) && \
    __has_builtin(__builtin_amdgcn_s_wait_asynccnt)
#define HAVE_ASYNC_LDS 1
#else
#define HAVE_ASYNC_LDS 0
#endif

typedef __attribute__((address_space(1))) v4i* gas_v4i_p;
typedef __attribute__((address_space(3))) v4i* las_v4i_p;

// Problem constants (from reference)
#define SRC_V 32000
#define TGT_V 32000
#define E_DIM 128
#define H_DIM 256
#define B_DIM 64
#define S_LEN 128
#define T_LEN 64
#define G4H   1024   // 4*H

// LDS B-tile row stride (halves): 40 -> 80B rows, bank-conflict-free b128 reads
#define BS_LD 40

// ---------------------------------------------------------------------------
// Fragment loaders following CDNA5 ISA 7.12.2 layouts.
// ---------------------------------------------------------------------------

// A-matrix 16x32 bf16 fragment from row-major [*, ld] at tile origin `tile`.
__device__ __forceinline__ v16bf load_frag_a(const bf16_t* tile, int ld, int lane) {
  const int r  = lane & 15;
  const int hi = lane >> 4;           // 0 or 1
  const bf16_t* p = tile + r * ld + hi * 8;
  v8bf lo = *(const v8bf*)(p);        // K = hi*8 .. +7
  v8bf hh = *(const v8bf*)(p + 16);   // K = hi*8+16 .. +7
  v16bf f;
#pragma unroll
  for (int i = 0; i < 8; ++i) { f[i] = lo[i]; f[i + 8] = hh[i]; }
  return f;
}

// Same, with row clamping for M-edge tiles.
__device__ __forceinline__ v16bf load_frag_a_clamp(const bf16_t* mat, int M, int ld,
                                                   int m0, int k0, int lane) {
  const int hi = lane >> 4;
  int r = m0 + (lane & 15);
  r = (r < M) ? r : (M - 1);
  const bf16_t* p = mat + r * ld + k0 + hi * 8;
  v8bf lo = *(const v8bf*)(p);
  v8bf hh = *(const v8bf*)(p + 16);
  v16bf f;
#pragma unroll
  for (int i = 0; i < 8; ++i) { f[i] = lo[i]; f[i + 8] = hh[i]; }
  return f;
}

// B-matrix 32x16 bf16 fragment consumed from the transpose W[N][K] in global.
__device__ __forceinline__ v16bf load_frag_bT(const bf16_t* bt, int ld, int lane) {
  const int n  = lane & 15;
  const int hi = lane >> 4;
  const bf16_t* p = bt + n * ld + hi * 8;
  v8bf lo = *(const v8bf*)(p);
  v8bf hh = *(const v8bf*)(p + 16);
  v16bf f;
#pragma unroll
  for (int i = 0; i < 8; ++i) { f[i] = lo[i]; f[i + 8] = hh[i]; }
  return f;
}

// B-matrix fragment from an LDS-staged tile with BS_LD row stride.
__device__ __forceinline__ v16bf load_frag_b_lds(const bf16_t* base, int lane) {
  const int n  = lane & 15;
  const int hi = lane >> 4;
  const bf16_t* p = base + n * BS_LD + hi * 8;
  v8bf lo = *(const v8bf*)(p);
  v8bf hh = *(const v8bf*)(p + 16);
  v16bf f;
#pragma unroll
  for (int i = 0; i < 8; ++i) { f[i] = lo[i]; f[i + 8] = hh[i]; }
  return f;
}

__device__ __forceinline__ float sigmoidf_(float x) {
  return 1.0f / (1.0f + __expf(-x));
}

// ---------------------------------------------------------------------------
// Kernel: fp32 -> bf16 conversion (weights)
// ---------------------------------------------------------------------------
__global__ void k_f32_to_bf16(const float* __restrict__ in, bf16_t* __restrict__ out, int n) {
  for (int i = blockIdx.x * blockDim.x + threadIdx.x; i < n; i += gridDim.x * blockDim.x)
    out[i] = (bf16_t)in[i];
}

// ---------------------------------------------------------------------------
// Kernel: embedding gather -> time-major bf16 activations X[l*B + b][E]
// ---------------------------------------------------------------------------
__global__ void k_gather_embed(const int* __restrict__ idx, int Lfull, int Luse,
                               const float* __restrict__ emb, bf16_t* __restrict__ X) {
  const int total = Luse * B_DIM * E_DIM;
  for (int i = blockIdx.x * blockDim.x + threadIdx.x; i < total; i += gridDim.x * blockDim.x) {
    const int e = i & (E_DIM - 1);
    const int m = i >> 7;            // l*B + b
    const int b = m & (B_DIM - 1);
    const int l = m >> 6;
    const int tok = idx[b * Lfull + l];
    X[i] = (bf16_t)emb[tok * E_DIM + e];
  }
}

// ---------------------------------------------------------------------------
// Kernel: input-gate pre-GEMM  Y[M][1024] = X[M][128] @ W[1024][128]^T + bias
// ---------------------------------------------------------------------------
__global__ void __launch_bounds__(256)
k_pregemm(const bf16_t* __restrict__ X, const bf16_t* __restrict__ W,
          const float* __restrict__ bias, float* __restrict__ Y, int M) {
  const int lane   = threadIdx.x & 31;
  const int wave   = threadIdx.x >> 5;
  const int m0     = blockIdx.x * 128 + wave * 16;
  const int n0     = blockIdx.y * 64;
  const int ln     = lane & 15;
  const int laneHi = lane >> 4;

  v8f acc[4];
#pragma unroll
  for (int g = 0; g < 4; ++g) {
    const float bv = bias[n0 + g * 16 + ln];
#pragma unroll
    for (int r = 0; r < 8; ++r) acc[g][r] = bv;
  }

#pragma unroll
  for (int kk = 0; kk < 4; ++kk) {
    const int k0 = kk * 32;
    v16bf a = load_frag_a_clamp(X, M, E_DIM, m0, k0, lane);
#pragma unroll
    for (int g = 0; g < 4; ++g) {
      __builtin_prefetch(W + (n0 + g * 16) * E_DIM + k0 + 32, 0, 1);
      v16bf b = load_frag_bT(W + (n0 + g * 16) * E_DIM + k0, E_DIM, lane);
      acc[g] = WMMA_BF16(a, b, acc[g]);
    }
  }

#pragma unroll
  for (int g = 0; g < 4; ++g) {
    const int col = n0 + g * 16 + ln;
#pragma unroll
    for (int r = 0; r < 8; ++r) {
      const int row = m0 + r + laneHi * 8;
      if (row < M) Y[row * G4H + col] = acc[g][r];
    }
  }
}

// ---------------------------------------------------------------------------
// Kernel: persistent single-WGP LSTM recurrence (128 enc + 63 dec steps).
// 1024 threads = 32 wave32s on one WGP; h in LDS (32 KB), c in registers;
// each wave owns all four gate tiles for its 16 hidden units so the cell
// update is wave-local (no gate exchange).
// ---------------------------------------------------------------------------
__global__ void __launch_bounds__(1024)
k_lstm_seq(const float* __restrict__ xg_enc, const float* __restrict__ xg_dec,
           const bf16_t* __restrict__ whh_enc, const bf16_t* __restrict__ whh_dec,
           bf16_t* __restrict__ hs_out) {
  __shared__ bf16_t h_lds[B_DIM * H_DIM];   // 32 KB

  const int tid    = threadIdx.x;
  const int lane   = tid & 31;
  const int wave   = tid >> 5;
  const int mh     = wave & 1;
  const int s      = wave >> 1;
  const int ln     = lane & 15;
  const int laneHi = lane >> 4;
  const int u      = s * 16 + ln;           // this thread's hidden unit

  for (int i = tid; i < B_DIM * H_DIM; i += 1024) h_lds[i] = (bf16_t)0.0f;

  float c[2][8];
#pragma unroll
  for (int mt = 0; mt < 2; ++mt)
#pragma unroll
    for (int r = 0; r < 8; ++r) c[mt][r] = 0.0f;

  __syncthreads();

  for (int t = 0; t < S_LEN + (T_LEN - 1); ++t) {
    const bool is_dec = (t >= S_LEN);
    const int  td     = t - S_LEN;
    const float*  xg  = is_dec ? (xg_dec + td * B_DIM * G4H) : (xg_enc + t * B_DIM * G4H);
    const bf16_t* whh = is_dec ? whh_dec : whh_enc;

    // ---- gates = Xg[t] (C-init) + h @ Whh^T (WMMA) ----
    v8f acc[2][4];
#pragma unroll
    for (int mt = 0; mt < 2; ++mt) {
      const int m0 = mh * 32 + mt * 16;
#pragma unroll
      for (int g = 0; g < 4; ++g) {
        const int col = g * 256 + u;
#pragma unroll
        for (int r = 0; r < 8; ++r)
          acc[mt][g][r] = xg[(m0 + r + laneHi * 8) * G4H + col];
      }
    }

#pragma unroll
    for (int kk = 0; kk < 8; ++kk) {
      const int k0 = kk * 32;
      v16bf a0 = load_frag_a(&h_lds[(mh * 32 + 0)  * H_DIM + k0], H_DIM, lane);
      v16bf a1 = load_frag_a(&h_lds[(mh * 32 + 16) * H_DIM + k0], H_DIM, lane);
#pragma unroll
      for (int g = 0; g < 4; ++g) {
        const int n0 = g * 256 + s * 16;
        v16bf b = load_frag_bT(whh + n0 * H_DIM + k0, H_DIM, lane);
        acc[0][g] = WMMA_BF16(a0, b, acc[0][g]);
        acc[1][g] = WMMA_BF16(a1, b, acc[1][g]);
      }
    }

    // ---- cell update, fully in-register ----
    float hval[2][8];
#pragma unroll
    for (int mt = 0; mt < 2; ++mt) {
#pragma unroll
      for (int r = 0; r < 8; ++r) {
        const float gi = acc[mt][0][r];
        const float gf = acc[mt][1][r];
        const float gg = acc[mt][2][r];
        const float go = acc[mt][3][r];
        const float cc = sigmoidf_(gf) * c[mt][r] + sigmoidf_(gi) * tanhf(gg);
        c[mt][r]    = cc;
        hval[mt][r] = sigmoidf_(go) * tanhf(cc);
      }
    }

    __syncthreads();   // all waves done reading old h
#pragma unroll
    for (int mt = 0; mt < 2; ++mt) {
#pragma unroll
      for (int r = 0; r < 8; ++r) {
        const int b = mh * 32 + mt * 16 + laneHi * 8 + r;
        const bf16_t hv = (bf16_t)hval[mt][r];
        h_lds[b * H_DIM + u] = hv;
        if (is_dec) hs_out[(td * B_DIM + b) * H_DIM + u] = hv;
      }
    }
    __syncthreads();   // new h visible to all waves
  }
}

// ---------------------------------------------------------------------------
// Kernel: zero the t=0 output slice (streaming/non-temporal stores)
// ---------------------------------------------------------------------------
__global__ void k_zero_t0(float* __restrict__ out) {
  const int total = B_DIM * TGT_V;
  for (int i = blockIdx.x * blockDim.x + threadIdx.x; i < total; i += gridDim.x * blockDim.x) {
    const int b = i / TGT_V;
    const int v = i - b * TGT_V;
    __builtin_nontemporal_store(0.0f, &out[(size_t)b * T_LEN * TGT_V + v]);
  }
}

// ---------------------------------------------------------------------------
// Kernel: final FC  logits[m][v] = hs[m] @ fc_W^T + fc_b, scattered into
// out[b][t+1][v] with m = t*64 + b.  M=4032, N=32000, K=256.
//
// fc_W tiles (128x32 bf16 = 8 KB) are double-buffer staged into LDS with
// gfx1250 async global->LDS loads (ASYNCcnt) so the 8 waves share one copy;
// LDS rows padded to 80 B for conflict-free ds_load_b128 fragment reads.
// Logit stores are non-temporal (524 MB write-once stream).
// ---------------------------------------------------------------------------
__global__ void __launch_bounds__(256)
k_fc(const bf16_t* __restrict__ HS, const bf16_t* __restrict__ W,
     const float* __restrict__ bias, float* __restrict__ out) {
  const int M = (T_LEN - 1) * B_DIM;   // 4032
  __shared__ bf16_t Bs[2][128 * BS_LD];  // 2 x 10 KB

  const int tid    = threadIdx.x;
  const int lane   = tid & 31;
  const int wave   = tid >> 5;
  const int m0     = blockIdx.x * 128 + wave * 16;
  const int n0     = blockIdx.y * 128;
  const int ln     = lane & 15;
  const int laneHi = lane >> 4;

  // Stage one 128(N) x 32(K) bf16 tile of W into Bs[buf]: 512 16B segments,
  // 2 per thread.
  auto stage = [&](int kk, int buf) {
#pragma unroll
    for (int j = 0; j < 2; ++j) {
      const int sg   = tid + 256 * j;
      const int row  = sg >> 2;        // 0..127
      const int part = sg & 3;         // 16B segment within the 64B row
      const bf16_t* gp = W + (size_t)(n0 + row) * H_DIM + kk * 32 + part * 8;
      bf16_t*       lp = &Bs[buf][row * BS_LD + part * 8];
#if HAVE_ASYNC_LDS
      __builtin_amdgcn_global_load_async_to_lds_b128(
          (gas_v4i_p)(uintptr_t)gp, (las_v4i_p)(uintptr_t)lp, 0, 0);
#else
      *(v8bf*)lp = *(const v8bf*)gp;
#endif
    }
  };

  v8f acc[8];
#pragma unroll
  for (int g = 0; g < 8; ++g) {
    const float bv = bias[n0 + g * 16 + ln];
#pragma unroll
    for (int r = 0; r < 8; ++r) acc[g][r] = bv;
  }

  stage(0, 0);

#pragma unroll
  for (int kk = 0; kk < 8; ++kk) {
    const int cur = kk & 1;
    if (kk + 1 < 8) stage(kk + 1, cur ^ 1);

#if HAVE_ASYNC_LDS
    // Our stage(kk) loads are complete once only stage(kk+1)'s 2 remain.
    if (kk + 1 < 8) __builtin_amdgcn_s_wait_asynccnt(2);
    else            __builtin_amdgcn_s_wait_asynccnt(0);
#endif
    __syncthreads();   // all lanes' stage(kk) writes visible

    const int k0 = kk * 32;
    v16bf a = load_frag_a_clamp(HS, M, H_DIM, m0, k0, lane);
    __builtin_prefetch(HS + m0 * H_DIM + k0 + 32, 0, 1);
#pragma unroll
    for (int g = 0; g < 8; ++g) {
      v16bf b = load_frag_b_lds(&Bs[cur][g * 16 * BS_LD], lane);
      acc[g] = WMMA_BF16(a, b, acc[g]);
    }
    __syncthreads();   // done reading Bs[cur] before it is restaged
  }

#pragma unroll
  for (int g = 0; g < 8; ++g) {
    const int col = n0 + g * 16 + ln;
#pragma unroll
    for (int r = 0; r < 8; ++r) {
      const int m = m0 + r + laneHi * 8;
      if (m < M) {
        const int b = m & (B_DIM - 1);
        const int t = m >> 6;
        __builtin_nontemporal_store(
            acc[g][r],
            &out[(size_t)b * T_LEN * TGT_V + (size_t)(t + 1) * TGT_V + col]);
      }
    }
  }
}

// ---------------------------------------------------------------------------
// Host launch
// ---------------------------------------------------------------------------
extern "C" void kernel_launch(void* const* d_in, const int* in_sizes, int n_in,
                              void* d_out, int out_size, void* d_ws, size_t ws_size,
                              hipStream_t stream) {
  const int*   src     = (const int*)  d_in[0];
  const int*   tgt     = (const int*)  d_in[1];
  const float* enc_emb = (const float*)d_in[2];
  const float* enc_Wih = (const float*)d_in[3];
  const float* enc_Whh = (const float*)d_in[4];
  const float* enc_b   = (const float*)d_in[5];
  const float* dec_emb = (const float*)d_in[6];
  const float* dec_Wih = (const float*)d_in[7];
  const float* dec_Whh = (const float*)d_in[8];
  const float* dec_b   = (const float*)d_in[9];
  const float* fc_W    = (const float*)d_in[10];
  const float* fc_b    = (const float*)d_in[11];
  float* out = (float*)d_out;

  // Workspace carve-up (256B aligned regions), ~73 MB total.
  char* ws = (char*)d_ws;
  auto alloc = [&](size_t bytes) -> char* {
    char* p = ws;
    ws += (bytes + 255) & ~(size_t)255;
    return p;
  };
  bf16_t* wih_e_bf = (bf16_t*)alloc((size_t)G4H * E_DIM * 2);
  bf16_t* whh_e_bf = (bf16_t*)alloc((size_t)G4H * H_DIM * 2);
  bf16_t* wih_d_bf = (bf16_t*)alloc((size_t)G4H * E_DIM * 2);
  bf16_t* whh_d_bf = (bf16_t*)alloc((size_t)G4H * H_DIM * 2);
  bf16_t* fcw_bf   = (bf16_t*)alloc((size_t)TGT_V * H_DIM * 2);
  bf16_t* x_enc    = (bf16_t*)alloc((size_t)S_LEN * B_DIM * E_DIM * 2);
  bf16_t* x_dec    = (bf16_t*)alloc((size_t)(T_LEN - 1) * B_DIM * E_DIM * 2);
  float*  xg_enc   = (float*) alloc((size_t)S_LEN * B_DIM * G4H * 4);
  float*  xg_dec   = (float*) alloc((size_t)(T_LEN - 1) * B_DIM * G4H * 4);
  bf16_t* hs       = (bf16_t*)alloc((size_t)(T_LEN - 1) * B_DIM * H_DIM * 2);

  const int M_enc = S_LEN * B_DIM;        // 8192
  const int M_dec = (T_LEN - 1) * B_DIM;  // 4032

  // 1) Weight conversion fp32 -> bf16
  k_f32_to_bf16<<<512, 256, 0, stream>>>(enc_Wih, wih_e_bf, G4H * E_DIM);
  k_f32_to_bf16<<<1024, 256, 0, stream>>>(enc_Whh, whh_e_bf, G4H * H_DIM);
  k_f32_to_bf16<<<512, 256, 0, stream>>>(dec_Wih, wih_d_bf, G4H * E_DIM);
  k_f32_to_bf16<<<1024, 256, 0, stream>>>(dec_Whh, whh_d_bf, G4H * H_DIM);
  k_f32_to_bf16<<<8192, 256, 0, stream>>>(fc_W, fcw_bf, TGT_V * H_DIM);

  // 2) Embedding gathers (time-major, bf16)
  k_gather_embed<<<2048, 256, 0, stream>>>(src, S_LEN, S_LEN, enc_emb, x_enc);
  k_gather_embed<<<2048, 256, 0, stream>>>(tgt, T_LEN, T_LEN - 1, dec_emb, x_dec);

  // 3) Input projections for all timesteps (bias folded in)
  {
    dim3 g_enc(M_enc / 128, G4H / 64);
    k_pregemm<<<g_enc, 256, 0, stream>>>(x_enc, wih_e_bf, enc_b, xg_enc, M_enc);
    dim3 g_dec((M_dec + 127) / 128, G4H / 64);
    k_pregemm<<<g_dec, 256, 0, stream>>>(x_dec, wih_d_bf, dec_b, xg_dec, M_dec);
  }

  // 4) Sequential recurrence: one persistent workgroup on one WGP
  k_lstm_seq<<<1, 1024, 0, stream>>>(xg_enc, xg_dec, whh_e_bf, whh_d_bf, hs);

  // 5) Output: zero t=0 slice, then HBM-bound WMMA FC into scattered layout
  k_zero_t0<<<4096, 256, 0, stream>>>(out);
  {
    dim3 g_fc((M_dec + 127) / 128, TGT_V / 128);
    k_fc<<<g_fc, 256, 0, stream>>>(hs, fcw_bf, fc_b, out);
  }
}
